// FlashAttention_34488587387519
// MI455X (gfx1250) — matrix operand order repaired and checked by reading, not compiled, not run on hardware
//
#include <hip/hip_runtime.h>

#define Bb 2
#define Hh 16
#define Ss 2048
#define Dd 128

typedef __attribute__((ext_vector_type(16))) __bf16 v16bf;
typedef __attribute__((ext_vector_type(8)))  float  v8f;

// Native f32 -> bf16 (gfx1250 bf16 cvt hardware; pairs fuse into v_cvt_pk_bf16_f32)
__device__ __forceinline__ __bf16 f2bf(float f) { return (__bf16)f; }

#define KSTR 136   // sK row stride (halves): 272B, mult of 16B -> b128-able, conflict-free
#define VSTR 40    // sVt row stride (halves): 80B, mult of 16B -> b128-able
#define PSTR 40    // sP row stride (halves): 80B

__global__ __launch_bounds__(128) void fa_fwd_kernel(
    const float* __restrict__ Q, const float* __restrict__ K,
    const float* __restrict__ V, float* __restrict__ O)
{
    __shared__ __bf16 sK [32 * KSTR];          // [key][d] row-major
    __shared__ __bf16 sVt[Dd * VSTR];          // [d][key] TRANSPOSED
    __shared__ __bf16 sP [4][16 * PSTR];

    const int bh    = blockIdx.y;
    const int qbase = blockIdx.x * 64;
    const int tid   = threadIdx.x;
    const int wave  = tid >> 5;
    const int lane  = tid & 31;
    const int half  = lane >> 4;
    const int l16   = lane & 15;
    const int qr0   = qbase + wave * 16;

    const size_t headoff = (size_t)bh * Ss * Dd;
    const float* Qh = Q + headoff;
    const float* Kh = K + headoff;
    const float* Vh = V + headoff;
    float*       Oh = O + headoff;

    // ---- Q fragments: A-matrix 16x32 bf16 layout, 4 K-chunks of 32 (D=128) ----
    v16bf aq[4];
    {
        const float* qrow = Qh + (size_t)(qr0 + l16) * Dd;
        #pragma unroll
        for (int c = 0; c < 4; ++c) {
            #pragma unroll
            for (int j = 0; j < 8; ++j) {
                aq[c][j]     = f2bf(qrow[c * 32 + half * 16 + j]);
                aq[c][8 + j] = f2bf(qrow[c * 32 + half * 16 + 8 + j]);
            }
        }
    }

    float mrow[8], lrow[8];
    v8f acc[8];
    #pragma unroll
    for (int r = 0; r < 8; ++r) { mrow[r] = -3.0e38f; lrow[r] = 0.0f; }
    #pragma unroll
    for (int dt = 0; dt < 8; ++dt)
        acc[dt] = (v8f){0.f, 0.f, 0.f, 0.f, 0.f, 0.f, 0.f, 0.f};

    const int ktend = qbase + 64;   // causal: only key blocks intersecting [0, qmax]
    for (int kt = 0; kt < ktend; kt += 32) {
        __syncthreads();
        // ---- K tile: 32 keys x 128 d, row-major, f32 -> bf16 ----
        {
            const float4* Kg = (const float4*)(Kh + (size_t)kt * Dd);
            #pragma unroll
            for (int i = 0; i < 8; ++i) {
                int idx = tid + 128 * i;        // 0..1023 float4 slots
                int row = idx >> 5;
                int c4  = idx & 31;
                float4 kv = Kg[row * 32 + c4];
                int o = row * KSTR + c4 * 4;
                sK[o + 0] = f2bf(kv.x); sK[o + 1] = f2bf(kv.y);
                sK[o + 2] = f2bf(kv.z); sK[o + 3] = f2bf(kv.w);
            }
        }
        // ---- V tile TRANSPOSED: thread t owns d-column t; coalesced b32 loads,
        //      4 packed bf16 (8B, merged ds_store) per 4 keys ----
        {
            const float* Vg = Vh + (size_t)kt * Dd + tid;
            #pragma unroll
            for (int i = 0; i < 8; ++i) {
                float v0 = Vg[(i * 4 + 0) * Dd];
                float v1 = Vg[(i * 4 + 1) * Dd];
                float v2 = Vg[(i * 4 + 2) * Dd];
                float v3 = Vg[(i * 4 + 3) * Dd];
                __bf16* p = &sVt[tid * VSTR + i * 4];
                p[0] = f2bf(v0); p[1] = f2bf(v1);
                p[2] = f2bf(v2); p[3] = f2bf(v3);
            }
        }
        if (kt + 32 < ktend) {   // global_prefetch_b8 of next tile
            __builtin_prefetch(Kh + (size_t)(kt + 32) * Dd + tid * 32, 0, 3);
            __builtin_prefetch(Vh + (size_t)(kt + 32) * Dd + tid * 32, 0, 3);
        }
        __syncthreads();

        // Wave-uniform skip of fully-masked tiles (no barriers below).
        if (kt > qr0 + 15) continue;
        const bool needMask = (kt + 31 > qr0);   // tile straddles the diagonal

        // ---- S = Q K^T for two 16-key subtiles (K-dim = D = 4x32 bf16 WMMA) ----
        float s[2][8];
        #pragma unroll
        for (int n = 0; n < 2; ++n) {
            v8f cs = (v8f){0.f, 0.f, 0.f, 0.f, 0.f, 0.f, 0.f, 0.f};
            const __bf16* krow = &sK[(n * 16 + l16) * KSTR];   // B col = key = l16
            #pragma unroll
            for (int c = 0; c < 4; ++c) {
                v16bf bk;
                #pragma unroll
                for (int e = 0; e < 16; ++e)                    // B: K = e + 16*half
                    bk[e] = krow[c * 32 + half * 16 + e];
                cs = __builtin_amdgcn_wmma_f32_16x16x32_bf16(
                        false, aq[c], false, bk, (short)0, cs, false, false);
            }
            if (needMask) {
                const int kabs = kt + n * 16 + l16;
                #pragma unroll
                for (int r = 0; r < 8; ++r) {
                    int qabs = qr0 + r + half * 8;              // C: M = r + 8*(lane>=16)
                    s[n][r] = (kabs > qabs) ? -3.4e38f : cs[r];
                }
            } else {
                #pragma unroll
                for (int r = 0; r < 8; ++r) s[n][r] = cs[r];
            }
        }

        // ---- online softmax over the 32 keys (both subtiles share one max) ----
        float scale[8];
        #pragma unroll
        for (int r = 0; r < 8; ++r) {
            float rm = fmaxf(s[0][r], s[1][r]);
            rm = fmaxf(rm, __shfl_xor(rm, 1, 32));
            rm = fmaxf(rm, __shfl_xor(rm, 2, 32));
            rm = fmaxf(rm, __shfl_xor(rm, 4, 32));
            rm = fmaxf(rm, __shfl_xor(rm, 8, 32));
            float mn = fmaxf(mrow[r], rm);
            scale[r] = __expf(mrow[r] - mn);
            mrow[r] = mn;
            float p0 = __expf(s[0][r] - mn);
            float p1 = __expf(s[1][r] - mn);
            s[0][r] = p0; s[1][r] = p1;
            float rs = p0 + p1;
            rs += __shfl_xor(rs, 1, 32);
            rs += __shfl_xor(rs, 2, 32);
            rs += __shfl_xor(rs, 4, 32);
            rs += __shfl_xor(rs, 8, 32);
            lrow[r] = lrow[r] * scale[r] + rs;
        }
        #pragma unroll
        for (int dt = 0; dt < 8; ++dt)
            #pragma unroll
            for (int r = 0; r < 8; ++r)
                acc[dt][r] *= scale[r];

        // ---- stash P (16x32 bf16) to per-wave LDS, reload as A fragment ----
        #pragma unroll
        for (int n = 0; n < 2; ++n)
            #pragma unroll
            for (int r = 0; r < 8; ++r)
                sP[wave][(r + half * 8) * PSTR + n * 16 + l16] = f2bf(s[n][r]);

        asm volatile("s_wait_dscnt 0" ::: "memory");

        v16bf ap;
        {
            const __bf16* prow = &sP[wave][l16 * PSTR];
            #pragma unroll
            for (int j = 0; j < 8; ++j) {
                ap[j]     = prow[half * 16 + j];
                ap[8 + j] = prow[half * 16 + 8 + j];
            }
        }

        // ---- O += P V : 8 WMMAs across D; B elem e -> key = e + 16*half,
        //      now contiguous in transposed sVt -> ds_load_b128 ----
        #pragma unroll
        for (int dt = 0; dt < 8; ++dt) {
            const __bf16* vrow = &sVt[(dt * 16 + l16) * VSTR + half * 16];
            v16bf bv;
            #pragma unroll
            for (int e = 0; e < 16; ++e)
                bv[e] = vrow[e];
            acc[dt] = __builtin_amdgcn_wmma_f32_16x16x32_bf16(
                        false, ap, false, bv, (short)0, acc[dt], false, false);
        }
    }

    // ---- epilogue: O / l ----
    #pragma unroll
    for (int r = 0; r < 8; ++r) {
        float inv = 1.0f / lrow[r];
        int row = qr0 + r + half * 8;
        float* orow = Oh + (size_t)row * Dd;
        #pragma unroll
        for (int dt = 0; dt < 8; ++dt)
            orow[dt * 16 + l16] = acc[dt][r] * inv;
    }
}

extern "C" void kernel_launch(void* const* d_in, const int* in_sizes, int n_in,
                              void* d_out, int out_size, void* d_ws, size_t ws_size,
                              hipStream_t stream) {
    const float* q = (const float*)d_in[0];
    const float* k = (const float*)d_in[1];
    const float* v = (const float*)d_in[2];
    // d_in[3] = causal mask; handled analytically in-kernel
    float* out = (float*)d_out;
    dim3 grid(Ss / 64, Bb * Hh);
    fa_fwd_kernel<<<grid, 128, 0, stream>>>(q, k, v, out);
}
